// Attention_27238682591514
// MI455X (gfx1250) — compile-verified, matrix operand order checked by reading, not run on hardware
//
#include <hip/hip_runtime.h>
#include <hip/hip_bf16.h>

typedef __attribute__((ext_vector_type(16))) _Float16 v16h;
typedef __attribute__((ext_vector_type(8)))  _Float16 v8h;
typedef __attribute__((ext_vector_type(4)))  _Float16 v4h;
typedef __attribute__((ext_vector_type(8)))  float    v8f;
typedef __attribute__((ext_vector_type(4)))  unsigned u32x4;
typedef __attribute__((ext_vector_type(8)))  int      i32x8;
typedef __attribute__((ext_vector_type(4)))  int      i32x4;

#define DEVI static __device__ __forceinline__

static constexpr int B_  = 2;
static constexpr int S_  = 2048;
static constexpr int D_  = 1024;
static constexpr int H_  = 16;
static constexpr int DH  = 64;
static constexpr int N3  = 3072;   // 3*D_

#if __has_builtin(__builtin_amdgcn_tensor_load_to_lds) && __has_builtin(__builtin_amdgcn_s_wait_tensorcnt)
#define HAS_TDM 1
#else
#define HAS_TDM 0
#endif

// WMMA as macro so modifier args stay literal immediates at the builtin call.
#define WMMA_F16(a, b, c) \
  __builtin_amdgcn_wmma_f32_16x16x32_f16(false, (a), false, (b), (short)0, (c), false, false)

// Per-lane 16-half fragment of a 16x32 operand: halves 0..7 = K koff..koff+7,
// halves 8..15 = K 16+koff..16+koff+7 (koff = (lane>>4)*8). Works for global or LDS.
DEVI v16h ldfrag(const _Float16* p) {
  v8h lo = *(const v8h*)p;
  v8h hi = *(const v8h*)(p + 16);
  return __builtin_shufflevector(lo, hi, 0,1,2,3,4,5,6,7,8,9,10,11,12,13,14,15);
}

// Low 32 bits of a generic pointer to LDS == LDS byte address (flat aperture rule).
DEVI unsigned lds_addr_of(const void* p) { return (unsigned)(uintptr_t)p; }

#if HAS_TDM
// Issue a 2-D TDM tile load: tile_rows rows x tile_k elements (2-byte data),
// row stride = row_stride elements, packed contiguously into LDS at lds_off.
// Toolchain here uses the 6-arg builtin: (g0 u32x4, g1 i32x8, g2 i32x4, g3 i32x4,
// extra i32x8, cpol i32); groups 2/3/extra are unused for a 2-D tile.
DEVI void tdm_load_2d(unsigned lds_off, const void* gsrc,
                      unsigned tensor_rows, unsigned row_stride,
                      unsigned tile_k, unsigned tile_rows) {
  unsigned long long ga = (unsigned long long)(uintptr_t)gsrc;
  u32x4 g0;
  g0[0] = 1u;                                              // count=1, is_restore=0
  g0[1] = lds_off;                                         // lds_addr [63:32]
  g0[2] = (unsigned)ga;                                    // global_addr lo
  g0[3] = (unsigned)((ga >> 32) & 0x01ffffffu)             // global_addr [56:32]
          | 0x80000000u;                                   // type=2 ("image")
  i32x8 g1;
  g1[0] = (int)(1u << 16);                                 // wg_mask=0, data_size=1 (2B)
  g1[1] = (int)((row_stride & 0xffffu) << 16);             // tensor_dim0[15:0] @ [63:48]
  g1[2] = (int)(((row_stride >> 16) & 0xffffu)             // tensor_dim0[31:16]
          | ((tensor_rows & 0xffffu) << 16));              // tensor_dim1[15:0]
  g1[3] = (int)(((tensor_rows >> 16) & 0xffffu)            // tensor_dim1[31:16]
          | ((tile_k & 0xffffu) << 16));                   // tile_dim0
  g1[4] = (int)(tile_rows & 0xffffu);                      // tile_dim1 (tile_dim2=0)
  g1[5] = (int)row_stride;                                 // tensor_dim0_stride lo32
  g1[6] = 0;                                               // stride0 hi / stride1 lo
  g1[7] = 0;
  i32x4 gz4 = {0, 0, 0, 0};
  i32x8 gz8 = {0, 0, 0, 0, 0, 0, 0, 0};
  __builtin_amdgcn_tensor_load_to_lds(g0, g1, gz4, gz4, gz8, 0);
}
#else
// Fallback: cooperative global->LDS copy of a rows x 32-half tile.
DEVI void coop_copy(_Float16* dst, const _Float16* src, int rows, int row_stride, int tid) {
  int chunks = rows * 4;                                   // 4 x v8h per row
  for (int c = tid; c < chunks; c += 256) {
    int r = c >> 2, s = c & 3;
    *(v8h*)(dst + r * 32 + s * 8) = *(const v8h*)(src + (size_t)r * row_stride + s * 8);
  }
}
#endif

// ---------------------------------------------------------------- prep kernels
__global__ void __launch_bounds__(256) cvt_x(const float* __restrict__ x,
                                             _Float16* __restrict__ xh, int n4) {
  int i = blockIdx.x * 256 + threadIdx.x;
  if (i >= n4) return;
  float4 f = ((const float4*)x)[i];
  v4h h;
  h[0] = (_Float16)f.x; h[1] = (_Float16)f.y;
  h[2] = (_Float16)f.z; h[3] = (_Float16)f.w;
  ((v4h*)xh)[i] = h;
}

__global__ void __launch_bounds__(256) cvt_w(const float* __restrict__ w,
                                             _Float16* __restrict__ wt, int n) {
  int i = blockIdx.x * 256 + threadIdx.x;     // coalesced read of W[k][n]
  if (i >= n) return;
  int k = i / N3;
  int c = i - k * N3;
  wt[(size_t)c * D_ + k] = (_Float16)w[i];    // Wt[n][k]
}

// ------------------------------------------------------------------- QKV GEMM
// Block (8 waves) computes a 128(M) x 64(N) tile; TDM stages the k-slices of A
// (128x32) and B (64x32) into double-buffered LDS; each wave does a 16x64 tile.
__global__ void __launch_bounds__(256) qkv_gemm(
    const _Float16* __restrict__ Xh, const _Float16* __restrict__ Wt,
    const float* __restrict__ bias,
    _Float16* __restrict__ Qh, _Float16* __restrict__ Kh, _Float16* __restrict__ Vt)
{
  __shared__ _Float16 Asm[2][128 * 32];   // 16 KB
  __shared__ _Float16 Bsm[2][64 * 32];    //  8 KB
  const int tid  = threadIdx.x;
  const int lane = tid & 31;
  const int l15  = lane & 15;
  const int hi   = lane >> 4;
  const int koff = hi * 8;
  const int wv   = tid >> 5;
  const int blkM = blockIdx.x / 48;
  const int blkN = blockIdx.x % 48;
  const int mBase = blkM * 128 + wv * 16;
  const int nBase = blkN * 64;

  const _Float16* Abase = Xh + (size_t)blkM * 128 * D_;
  const _Float16* Bbase = Wt + (size_t)nBase * D_;
  constexpr int NT = D_ / 32;

#if HAS_TDM
  if (wv == 0)      tdm_load_2d(lds_addr_of(&Bsm[0][0]), Bbase,  64, D_, 32,  64);
  else if (wv == 1) tdm_load_2d(lds_addr_of(&Asm[0][0]), Abase, 128, D_, 32, 128);
#else
  coop_copy(&Bsm[0][0], Bbase, 64, D_, tid);
  coop_copy(&Asm[0][0], Abase, 128, D_, tid);
#endif

  v8f acc[4] = {};
  for (int i = 0; i < NT; ++i) {
    const int cur = i & 1;
#if HAS_TDM
    if (wv == 0) {
      if (i + 1 < NT) {
        tdm_load_2d(lds_addr_of(&Bsm[cur ^ 1][0]), Bbase + (i + 1) * 32, 64, D_, 32, 64);
        __builtin_amdgcn_s_wait_tensorcnt(1);
      } else {
        __builtin_amdgcn_s_wait_tensorcnt(0);
      }
    } else if (wv == 1) {
      if (i + 1 < NT) {
        tdm_load_2d(lds_addr_of(&Asm[cur ^ 1][0]), Abase + (i + 1) * 32, 128, D_, 32, 128);
        __builtin_amdgcn_s_wait_tensorcnt(1);
      } else {
        __builtin_amdgcn_s_wait_tensorcnt(0);
      }
    }
#endif
    __syncthreads();                         // staged buffers visible to all waves

    const _Float16* Ab = &Asm[cur][0] + (wv * 16 + l15) * 32 + koff;
    const _Float16* Bb = &Bsm[cur][0];
    v16h a  = ldfrag(Ab);
    v16h b0 = ldfrag(Bb + (0 * 16 + l15) * 32 + koff);
    v16h b1 = ldfrag(Bb + (1 * 16 + l15) * 32 + koff);
    v16h b2 = ldfrag(Bb + (2 * 16 + l15) * 32 + koff);
    v16h b3 = ldfrag(Bb + (3 * 16 + l15) * 32 + koff);
    acc[0] = WMMA_F16(a, b0, acc[0]);
    acc[1] = WMMA_F16(a, b1, acc[1]);
    acc[2] = WMMA_F16(a, b2, acc[2]);
    acc[3] = WMMA_F16(a, b3, acc[3]);

    __syncthreads();                         // all reads done before next overwrite
#if !HAS_TDM
    if (i + 1 < NT) {
      coop_copy(&Bsm[cur ^ 1][0], Bbase + (i + 1) * 32, 64, D_, tid);
      coop_copy(&Asm[cur ^ 1][0], Abase + (i + 1) * 32, 128, D_, tid);
    }
#endif
  }

#pragma unroll
  for (int t = 0; t < 4; ++t) {
    const int col = nBase + t * 16 + l15;   // 0..3071
    const float bv = bias[col];
    const int region = col >> 10;           // 0=Q 1=K 2=V
    const int cc = col & (D_ - 1);
    const int h  = cc >> 6;
    const int dh = cc & 63;
    if (region == 0) {
#pragma unroll
      for (int r = 0; r < 8; ++r) {
        int row = mBase + r + 8 * hi;
        int bb = row >> 11, s = row & (S_ - 1);
        Qh[(((size_t)(bb * H_ + h)) * S_ + s) * DH + dh] =
            (_Float16)((acc[t][r] + bv) * 0.125f);
      }
    } else if (region == 1) {
#pragma unroll
      for (int r = 0; r < 8; ++r) {
        int row = mBase + r + 8 * hi;
        int bb = row >> 11, s = row & (S_ - 1);
        Kh[(((size_t)(bb * H_ + h)) * S_ + s) * DH + dh] = (_Float16)(acc[t][r] + bv);
      }
    } else {
#pragma unroll
      for (int r = 0; r < 8; ++r) {
        int row = mBase + r + 8 * hi;
        int bb = row >> 11, s = row & (S_ - 1);
        Vt[(((size_t)(bb * H_ + h)) * DH + dh) * S_ + s] = (_Float16)(acc[t][r] + bv);
      }
    }
  }
}

// ---------------------------------------------------------------- attention
// One wave per 16-row Q block of one (b,h). Flash-attention over KV in 32-col steps.
__global__ void __launch_bounds__(256) attn_fwd(
    const _Float16* __restrict__ Qh, const _Float16* __restrict__ Kh,
    const _Float16* __restrict__ Vt, float* __restrict__ out)
{
  __shared__ _Float16 Pbuf[8][16 * 32];     // per-wave 16x32 P tile (8 KB total)
  const int lane = threadIdx.x & 31;
  const int l15  = lane & 15;
  const int hi   = lane >> 4;
  const int koff = hi * 8;
  const int wv   = threadIdx.x >> 5;
  const int w    = blockIdx.x * 8 + wv;
  const int qBlk = w & 127;                 // 128 q-blocks of 16 rows
  const int bh   = w >> 7;                  // 0..31
  const _Float16* Qb = Qh + (size_t)bh * S_ * DH;
  const _Float16* Kb = Kh + (size_t)bh * S_ * DH;
  const _Float16* Vb = Vt + (size_t)bh * DH * S_;
  _Float16* P = &Pbuf[wv][0];

  const _Float16* qrow = Qb + (size_t)(qBlk * 16 + l15) * DH;
  v16h aq0 = ldfrag(qrow + 0  + koff);
  v16h aq1 = ldfrag(qrow + 32 + koff);

  v8f o0 = {}, o1 = {}, o2 = {}, o3 = {};
  float mrow[8], lsum[8];
#pragma unroll
  for (int r = 0; r < 8; ++r) { mrow[r] = -3.0e38f; lsum[r] = 0.f; }

  for (int j0 = 0; j0 < S_; j0 += 32) {
    if (j0 + 32 < S_) {                     // prefetch next KV block
      __builtin_prefetch(Kb + (size_t)(j0 + 32 + l15) * DH, 0, 0);
      __builtin_prefetch(Vb + (size_t)l15 * S_ + j0 + 32, 0, 0);
    }
    // ---- scores: S[16x32] = Q . K^T  (scale folded into Q)
    v8f s0 = {}, s1 = {};
    {
      const _Float16* kr0 = Kb + (size_t)(j0 + l15) * DH;
      const _Float16* kr1 = Kb + (size_t)(j0 + 16 + l15) * DH;
      s0 = WMMA_F16(aq0, ldfrag(kr0 + 0  + koff), s0);
      s0 = WMMA_F16(aq1, ldfrag(kr0 + 32 + koff), s0);
      s1 = WMMA_F16(aq0, ldfrag(kr1 + 0  + koff), s1);
      s1 = WMMA_F16(aq1, ldfrag(kr1 + 32 + koff), s1);
    }
    // ---- online softmax (rows live in 16-lane groups of the C layout)
#pragma unroll
    for (int r = 0; r < 8; ++r) {
      float v = fmaxf(s0[r], s1[r]);
      v = fmaxf(v, __shfl_xor(v, 8, 32));
      v = fmaxf(v, __shfl_xor(v, 4, 32));
      v = fmaxf(v, __shfl_xor(v, 2, 32));
      v = fmaxf(v, __shfl_xor(v, 1, 32));
      float mn = fmaxf(mrow[r], v);
      float al = __expf(mrow[r] - mn);
      float p0 = __expf(s0[r] - mn);
      float p1 = __expf(s1[r] - mn);
      s0[r] = p0; s1[r] = p1;
      float rs = p0 + p1;
      rs += __shfl_xor(rs, 8, 32);
      rs += __shfl_xor(rs, 4, 32);
      rs += __shfl_xor(rs, 2, 32);
      rs += __shfl_xor(rs, 1, 32);
      lsum[r] = lsum[r] * al + rs;
      o0[r] *= al; o1[r] *= al; o2[r] *= al; o3[r] *= al;
      mrow[r] = mn;
    }
    // ---- C-layout -> A-layout for P via wave-private LDS tile
#pragma unroll
    for (int r = 0; r < 8; ++r) {
      P[(r + 8 * hi) * 32 + l15]      = (_Float16)s0[r];
      P[(r + 8 * hi) * 32 + 16 + l15] = (_Float16)s1[r];
    }
    asm volatile("s_wait_dscnt 0" ::: "memory");
    v16h pa = ldfrag(P + l15 * 32 + koff);
    // ---- O += P . V   (Vt is [dh][s]: B loads are K(=s)-contiguous)
    o0 = WMMA_F16(pa, ldfrag(Vb + (size_t)(0  + l15) * S_ + j0 + koff), o0);
    o1 = WMMA_F16(pa, ldfrag(Vb + (size_t)(16 + l15) * S_ + j0 + koff), o1);
    o2 = WMMA_F16(pa, ldfrag(Vb + (size_t)(32 + l15) * S_ + j0 + koff), o2);
    o3 = WMMA_F16(pa, ldfrag(Vb + (size_t)(48 + l15) * S_ + j0 + koff), o3);
  }

  // ---- normalize and write fp32 output [b, s, h*64 + dh]
  const int b = bh >> 4, h = bh & 15;
#pragma unroll
  for (int r = 0; r < 8; ++r) {
    int s = qBlk * 16 + r + 8 * hi;
    float inv = 1.0f / lsum[r];
    size_t base = ((size_t)(b * S_ + s)) * D_ + h * DH + l15;
    out[base + 0]  = o0[r] * inv;
    out[base + 16] = o1[r] * inv;
    out[base + 32] = o2[r] * inv;
    out[base + 48] = o3[r] * inv;
  }
}

// ------------------------------------------------------------------- launcher
extern "C" void kernel_launch(void* const* d_in, const int* in_sizes, int n_in,
                              void* d_out, int out_size, void* d_ws, size_t ws_size,
                              hipStream_t stream) {
  const float* x    = (const float*)d_in[0];   // [2,2048,1024]
  const float* Wq   = (const float*)d_in[1];   // [1024,3072]
  const float* bias = (const float*)d_in[2];   // [3072]
  float* out = (float*)d_out;

  char* ws = (char*)d_ws;                      // needs ~38 MB
  _Float16* Xh = (_Float16*)(ws + (size_t) 0 * 1024 * 1024);  // 8 MB
  _Float16* Wt = (_Float16*)(ws + (size_t) 8 * 1024 * 1024);  // 6 MB
  _Float16* Qh = (_Float16*)(ws + (size_t)14 * 1024 * 1024);  // 8 MB
  _Float16* Kh = (_Float16*)(ws + (size_t)22 * 1024 * 1024);  // 8 MB
  _Float16* Vt = (_Float16*)(ws + (size_t)30 * 1024 * 1024);  // 8 MB

  {
    int n4 = (B_ * S_ * D_) / 4;               // 1048576
    cvt_x<<<dim3(n4 / 256), dim3(256), 0, stream>>>(x, Xh, n4);
  }
  {
    int n = D_ * N3;                           // 3145728
    cvt_w<<<dim3(n / 256), dim3(256), 0, stream>>>(Wq, Wt, n);
  }
  {
    // blocks: (4096/128) x (3072/64) = 32 x 48 = 1536, 8 waves each
    qkv_gemm<<<dim3(1536), dim3(256), 0, stream>>>(Xh, Wt, bias, Qh, Kh, Vt);
  }
  {
    // waves = 2*16*(2048/16) = 4096 -> 512 blocks of 8 waves
    attn_fwd<<<dim3(512), dim3(256), 0, stream>>>(Qh, Kh, Vt, out);
  }
}
